// VGAE_51616916963855
// MI455X (gfx1250) — matrix operand order repaired and checked by reading, not compiled, not run on hardware
//
#include <hip/hip_runtime.h>
#include <hip/hip_bf16.h>

typedef __bf16 bf16;
typedef bf16  v16bf __attribute__((ext_vector_type(16)));
typedef float v8f   __attribute__((ext_vector_type(8)));
typedef unsigned int u32x4 __attribute__((ext_vector_type(4)));
typedef int          i32x8 __attribute__((ext_vector_type(8)));
typedef int          i32x4 __attribute__((ext_vector_type(4)));

#define NN 1024
#define HIDD 128
#define LATD 64

#if __has_builtin(__builtin_amdgcn_tensor_load_to_lds) && \
    __has_builtin(__builtin_amdgcn_s_wait_tensorcnt)
#define HAVE_TDM 1
#endif

// ---------- degree -> dis = clip(deg,1e-9)^-0.5 ----------
__global__ void k_degree(const float* __restrict__ adj, float* __restrict__ dis) {
    __shared__ float red[256];
    const int i = blockIdx.x;
    float s = 0.f;
    for (int j = threadIdx.x; j < NN; j += 256) s += adj[i * NN + j];
    red[threadIdx.x] = s;
    __syncthreads();
    for (int w = 128; w > 0; w >>= 1) {
        if (threadIdx.x < w) red[threadIdx.x] += red[threadIdx.x + w];
        __syncthreads();
    }
    if (threadIdx.x == 0) dis[i] = rsqrtf(fmaxf(red[0], 1e-9f));
}

// ---------- T[i,c] = sum_k adj[i,k] * dis[k] * S[k,c]   (128 cols) ----------
__global__ void k_adjgemm(const float* __restrict__ adj, const float* __restrict__ dis,
                          const float* __restrict__ S, float* __restrict__ out) {
    const int gid = blockIdx.x * 256 + threadIdx.x;
    const int i = gid >> 7, c = gid & 127;
    float acc = 0.f;
    for (int k = 0; k < NN; ++k) {
        const float a = adj[i * NN + k];      // wave-uniform (all lanes share i)
        if (a != 0.f) acc = fmaf(a * dis[k], S[k * HIDD + c], acc);
    }
    out[gid] = acc;
}

// ---------- h = relu(dis_i * (T @ W1) + b1) ----------
__global__ void k_lin_relu(const float* __restrict__ T, const float* __restrict__ dis,
                           const float* __restrict__ W, const float* __restrict__ b,
                           float* __restrict__ h) {
    const int gid = blockIdx.x * 256 + threadIdx.x;
    const int i = gid >> 7, l = gid & 127;
    float acc = 0.f;
    for (int k = 0; k < HIDD; ++k) acc = fmaf(T[i * HIDD + k], W[k * HIDD + l], acc);
    h[gid] = fmaxf(fmaf(dis[i], acc, b[l]), 0.f);
}

// ---------- mu, logvar, z = mu + eps*exp(0.5*lv) ----------
__global__ void k_head(const float* __restrict__ T2, const float* __restrict__ dis,
                       const float* __restrict__ Wmu, const float* __restrict__ bmu,
                       const float* __restrict__ Wlv, const float* __restrict__ blv,
                       const float* __restrict__ eps,
                       float* __restrict__ mu_out, float* __restrict__ lv_out,
                       float* __restrict__ zf, bf16* __restrict__ zb) {
    const int gid = blockIdx.x * 256 + threadIdx.x;
    const int i = gid >> 6, l = gid & 63;
    float am = 0.f, al = 0.f;
    for (int k = 0; k < HIDD; ++k) {
        const float t = T2[i * HIDD + k];
        am = fmaf(t, Wmu[k * LATD + l], am);
        al = fmaf(t, Wlv[k * LATD + l], al);
    }
    const float m  = fmaf(dis[i], am, bmu[l]);
    const float lv = fmaf(dis[i], al, blv[l]);
    mu_out[gid] = m;
    lv_out[gid] = lv;
    const float z = fmaf(eps[gid], expf(0.5f * lv), m);
    zf[gid] = z;
    zb[gid] = (bf16)z;
}

// ---------- U = z@(Wa-Wc),  V = z@(Wb+Wc)+br1 ----------
__global__ void k_uv(const float* __restrict__ zf, const float* __restrict__ Wr1,
                     const float* __restrict__ br1,
                     float* __restrict__ U, float* __restrict__ V) {
    const int gid = blockIdx.x * 256 + threadIdx.x;
    const int i = gid >> 6, l = gid & 63;
    float u = 0.f, v = 0.f;
    for (int k = 0; k < LATD; ++k) {
        const float zk = zf[i * LATD + k];
        const float wa = Wr1[(k)       * LATD + l];
        const float wb = Wr1[(64 + k)  * LATD + l];
        const float wc = Wr1[(128 + k) * LATD + l];
        u = fmaf(zk, wa - wc, u);
        v = fmaf(zk, wb + wc, v);
    }
    U[gid] = u;
    V[gid] = v + br1[l];
}

// ---------- decoder ----------
// scores[i,j] = relu(U[i]+V[j]+ bilinear(z_i, z_j; Wd)) . Wr2 + br2
#define MT_STRIDE 72   // bf16: 144B rows, 16B aligned, conflict-free A-frag reads
#define V_STRIDE  68   // f32: pad 4 dwords per 64 -> bank = 4*j + l (conflict-free)

__global__ __launch_bounds__(256) void k_decoder(
    const float* __restrict__ zf, const bf16* __restrict__ zb,
    const float* __restrict__ Wr1, const float* __restrict__ Wr2,
    const float* __restrict__ br2, const float* __restrict__ U,
    const float* __restrict__ V, float* __restrict__ scores) {
    __shared__ bf16  MT[64 * MT_STRIDE];                 // MT[l][k] = z_i[k]*Wd[k][l]
    __shared__ float Us[64], W2s[64];
    __shared__ __align__(128) float Vbuf[128 * V_STRIDE];

    const int i = blockIdx.x;
    const int tid = threadIdx.x;

    for (int idx = tid; idx < 64 * 64; idx += 256) {
        const int k = idx & 63, l = idx >> 6;
        MT[l * MT_STRIDE + k] = (bf16)(zf[i * 64 + k] * Wr1[(192 + k) * 64 + l]);
    }
    if (tid < 64) { Us[tid] = U[i * 64 + tid]; W2s[tid] = Wr2[tid]; }
    __syncthreads();   // publish MT before hoisting A fragments

    const int wave = tid >> 5, lane = tid & 31;
    const int half = lane >> 4, ln = lane & 15;
    const float b2 = br2[0];

    // ---- hoist the 8 chunk-invariant A fragments (rows of M_i^T) into VGPRs ----
    v16bf afr[2][4];
#pragma unroll
    for (int s = 0; s < 2; ++s)
#pragma unroll
        for (int mt = 0; mt < 4; ++mt) {
            const bf16* rp = &MT[(mt * 16 + ln) * MT_STRIDE + s * 32 + half * 16];
#pragma unroll
            for (int e = 0; e < 8; ++e) {
                afr[s][mt][e]     = rp[e];
                afr[s][mt][8 + e] = rp[8 + e];
            }
        }

    for (int chunk = 0; chunk < 8; ++chunk) {
        // ---- kick off async TDM of V[chunk*128 .. +128) into padded LDS rows ----
#ifdef HAVE_TDM
        if (wave == 0) {
            const unsigned long long ga =
                (unsigned long long)(uintptr_t)(V + chunk * 128 * 64);
            u32x4 g0;
            g0[0] = 1u;                                    // count=1 (valid user D#)
            g0[1] = (unsigned)(uintptr_t)(void*)Vbuf;      // lds_addr (low 32 = LDS off)
            g0[2] = (unsigned)(ga & 0xffffffffu);          // global_addr[31:0]
            g0[3] = (unsigned)((ga >> 32) & 0x1ffffffu)    // global_addr[56:32]
                    | (2u << 30);                          // type = 2 (image)
            i32x8 g1;
            g1[0] = (int)((2u << 16)       // data_size = 2 (4 bytes)
                        | (1u << 20)       // pad_enable
                        | (5u << 22)       // pad_interval: every 64 dwords
                        | (3u << 25));     // pad_amount: 4 dwords
            g1[1] = (int)((8192u & 0xffffu) << 16);        // tensor_dim0[15:0]
            g1[2] = (int)(((8192u >> 16) & 0xffffu)        // tensor_dim0[31:16]
                        | (1u << 16));                     // tensor_dim1 = 1
            g1[3] = (int)((8192u & 0xffffu) << 16);        // tile_dim0 = 8192
            g1[4] = 0;                                     // tile_dim1/2 unused
            g1[5] = (int)8192u;                            // tensor_dim0_stride
            g1[6] = 0;
            g1[7] = 0;
            const i32x4 gz4 = {0, 0, 0, 0};
            const i32x8 gz8 = {0, 0, 0, 0, 0, 0, 0, 0};
            __builtin_amdgcn_tensor_load_to_lds(g0, g1, gz4, gz4, gz8, 0);
            // no wait here: DMA overlaps with the WMMA phase below
        }
#else
        for (int idx = tid; idx < 128 * 64; idx += 256) {
            const int jl = idx >> 6, l = idx & 63;
            Vbuf[jl * V_STRIDE + l] = V[(chunk * 128 + jl) * 64 + l];
        }
#endif

        const int jbase = chunk * 128 + wave * 16;
        const int j = jbase + ln;
        const int jl = wave * 16 + ln;          // row within Vbuf

        if (j + 128 < NN)                       // prefetch next chunk's B strip
            __builtin_prefetch(zb + (j + 128) * 64, 0, 3);

        v8f c[4] = {v8f{}, v8f{}, v8f{}, v8f{}};

#pragma unroll
        for (int s = 0; s < 2; ++s) {
            // B fragment (K x N = 32 x 16): lane = column j, halves split K
            v16bf bfrag;
            const bf16* zr = zb + j * 64 + s * 32 + half * 16;
#pragma unroll
            for (int e = 0; e < 16; ++e) bfrag[e] = zr[e];

#pragma unroll
            for (int mt = 0; mt < 4; ++mt)
                c[mt] = __builtin_amdgcn_wmma_f32_16x16x32_bf16(
                    false, afr[s][mt], false, bfrag, (short)0, c[mt], false, false);
        }

#ifdef HAVE_TDM
        if (wave == 0) __builtin_amdgcn_s_wait_tensorcnt(0);
#endif
        __syncthreads();                        // Vbuf ready for everyone

        // epilogue: t = bilinear + U[i,l] + V[j,l]; acc += relu(t)*Wr2[l]
        float acc = 0.f;
#pragma unroll
        for (int mt = 0; mt < 4; ++mt) {
#pragma unroll
            for (int v = 0; v < 8; ++v) {
                const int l = mt * 16 + half * 8 + v;
                const float t = c[mt][v] + Us[l] + Vbuf[jl * V_STRIDE + l];
                acc = fmaf(fmaxf(t, 0.f), W2s[l], acc);
            }
        }
        acc += __shfl_xor(acc, 16, 32);
        if (lane < 16) scores[i * NN + jbase + lane] = acc + b2;

        __syncthreads();                        // Vbuf reused next chunk
    }
}

extern "C" void kernel_launch(void* const* d_in, const int* in_sizes, int n_in,
                              void* d_out, int out_size, void* d_ws, size_t ws_size,
                              hipStream_t stream) {
    const float* x   = (const float*)d_in[0];
    const float* adj = (const float*)d_in[1];
    const float* eps = (const float*)d_in[2];
    const float* W1  = (const float*)d_in[3];
    const float* b1  = (const float*)d_in[4];
    const float* Wmu = (const float*)d_in[5];
    const float* bmu = (const float*)d_in[6];
    const float* Wlv = (const float*)d_in[7];
    const float* blv = (const float*)d_in[8];
    const float* Wr1 = (const float*)d_in[9];
    const float* br1 = (const float*)d_in[10];
    const float* Wr2 = (const float*)d_in[11];
    const float* br2 = (const float*)d_in[12];

    float* out    = (float*)d_out;
    float* scores = out;                    // [1024*1024]
    float* mu_out = out + NN * NN;          // [1024*64]
    float* lv_out = mu_out + NN * LATD;     // [1024*64]

    float* ws  = (float*)d_ws;
    float* dis = ws;                        // 1024
    float* T   = dis + 1024;                // 1024*128
    float* hb  = T + NN * HIDD;             // 1024*128
    float* zfb = hb + NN * HIDD;            // 1024*64
    float* U   = zfb + NN * LATD;           // 1024*64
    float* V   = U + NN * LATD;             // 1024*64
    bf16*  zb  = (bf16*)(V + NN * LATD);    // 1024*64 bf16

    k_degree  <<<NN, 256, 0, stream>>>(adj, dis);
    k_adjgemm <<<NN * HIDD / 256, 256, 0, stream>>>(adj, dis, x, T);
    k_lin_relu<<<NN * HIDD / 256, 256, 0, stream>>>(T, dis, W1, b1, hb);
    k_adjgemm <<<NN * HIDD / 256, 256, 0, stream>>>(adj, dis, hb, T);
    k_head    <<<NN * LATD / 256, 256, 0, stream>>>(T, dis, Wmu, bmu, Wlv, blv, eps,
                                                    mu_out, lv_out, zfb, zb);
    k_uv      <<<NN * LATD / 256, 256, 0, stream>>>(zfb, Wr1, br1, U, V);
    k_decoder <<<NN, 256, 0, stream>>>(zfb, zb, Wr1, Wr2, br2, U, V, scores);
}
